// ScalarDotAttention_38319698215657
// MI455X (gfx1250) — matrix operand order repaired and checked by reading, not compiled, not run on hardware
//
#include <hip/hip_runtime.h>

typedef __attribute__((ext_vector_type(16))) __bf16 v16bf;
typedef __attribute__((ext_vector_type(8)))  float  v8f;
typedef unsigned int u32x4 __attribute__((ext_vector_type(4)));
typedef int          i32x8 __attribute__((ext_vector_type(8)));
typedef int          i32x4 __attribute__((ext_vector_type(4)));

#define SEQ_L     2048
#define HEAD_D    128
#define Q_TILE    64
#define K_TILE    64
#define N_KCHUNK  (SEQ_L / K_TILE)
#define BLOCK_T   128
#define QK_STRIDE 136   // 128 data halves + 8 pad  (= 64 DW + 4 DW pad per row)
#define V_STRIDE  72    // 64 data halves + 8 pad   (= 32 DW + 4 DW pad per row)
#define P_STRIDE  72

#if defined(__has_builtin)
#  if __has_builtin(__builtin_amdgcn_tensor_load_to_lds) && __has_builtin(__builtin_amdgcn_s_wait_tensorcnt)
#    define HAVE_TDM 1
#  endif
#endif
#ifndef HAVE_TDM
#  define HAVE_TDM 0
#endif

static __device__ __forceinline__ unsigned short f2bf(float x) {
  __bf16 h = (__bf16)x;
  return __builtin_bit_cast(unsigned short, h);
}

static __device__ __forceinline__ void pack4_store(unsigned short* p, float4 f) {
  unsigned long long pk =
      (unsigned long long)f2bf(f.x)         |
      ((unsigned long long)f2bf(f.y) << 16) |
      ((unsigned long long)f2bf(f.z) << 32) |
      ((unsigned long long)f2bf(f.w) << 48);
  *(unsigned long long*)p = pk;
}

// A-operand fragment (16x32 bf16): lane row = lane&15, K chunks [hi*8,+8) and [16+hi*8,+8)
static __device__ __forceinline__ v16bf load_frag_a(const unsigned short* rowBase, int hi) {
  union { v16bf v; uint4 q[2]; } f;
  f.q[0] = *(const uint4*)(rowBase + hi * 8);
  f.q[1] = *(const uint4*)(rowBase + 16 + hi * 8);
  return f.v;
}
// B-operand fragment (32x16 bf16): lane column = lane&15, K chunks [hi*8,+8) and [16+hi*8,+8)
static __device__ __forceinline__ v16bf load_frag_b(const unsigned short* colBase, int hi) {
  union { v16bf v; uint4 q[2]; } f;
  f.q[0] = *(const uint4*)(colBase + hi * 8);
  f.q[1] = *(const uint4*)(colBase + 16 + hi * 8);
  return f.v;
}

static __device__ __forceinline__ unsigned lds_addr_of(const void* p) {
  // generic pointers to LDS carry the byte offset in the low 32 bits (aperture scheme)
  return (unsigned)(unsigned long long)(size_t)p;
}

#if HAVE_TDM
// Build a 2D-tile D# per CDNA5 ISA and issue TENSOR_LOAD_TO_LDS.
// data_size = 2 bytes; pad_enable reproduces the padded LDS row strides.
// 6-arg builtin form: (uint32x4 g0, int32x8 g1, int32x4 g2, int32x4 g3, int32x8 g4, i32 cpol)
static __device__ __forceinline__ void tdm_load_2d(unsigned ldsAddr, unsigned long long gaddr,
                                                   unsigned td0, unsigned td1,
                                                   unsigned tile0, unsigned tile1,
                                                   unsigned long long stride0,
                                                   unsigned padIntervalCode, unsigned padAmountCode) {
  u32x4 g0;
  g0[0] = 1u;                                                  // count=1, user descriptor
  g0[1] = ldsAddr;                                             // lds_addr [63:32]
  g0[2] = (unsigned)(gaddr & 0xffffffffull);                   // global_addr [95:64]
  g0[3] = (unsigned)((gaddr >> 32) & 0x1ffffffull) | (2u << 30); // addr[56:32] | type=2
  i32x8 g1;
  g1[0] = (int)((1u << 16)                                     // data_size = 2B
              | (1u << 20)                                     // pad_enable
              | (padIntervalCode << 22) | (padAmountCode << 25));
  g1[1] = (int)((td0 & 0xffffu) << 16);                        // tensor_dim0[15:0] @ bits 63:48
  g1[2] = (int)(((td0 >> 16) & 0xffffu) | ((td1 & 0xffffu) << 16));
  g1[3] = (int)(((td1 >> 16) & 0xffffu) | ((tile0 & 0xffffu) << 16));
  g1[4] = (int)(tile1 & 0xffffu);                              // tile_dim1; tile_dim2=0
  g1[5] = (int)(unsigned)(stride0 & 0xffffffffull);            // tensor_dim0_stride[31:0]
  g1[6] = (int)(unsigned)((stride0 >> 32) & 0xffffu);          // stride[47:32]; dim1_stride unused
  g1[7] = 0;
  i32x4 z4 = {0, 0, 0, 0};
  i32x8 z8 = {0, 0, 0, 0, 0, 0, 0, 0};
  __builtin_amdgcn_tensor_load_to_lds(g0, g1, z4, z4, z8, 0);
}
#endif

// ---- fetch one K/V chunk into an LDS buffer (TDM if available, else plain bf16 copy) ----
// isIssuer must be a wave-uniform SGPR condition: tensor ops ignore EXEC, so the guard
// must be a scalar branch (readfirstlane), not an EXEC-mask region.
static __device__ __forceinline__ void issue_chunk_load(const unsigned short* Kchunk,
                                                        const unsigned short* Vchunk,
                                                        unsigned short* sKbuf,
                                                        unsigned short* sVbuf,
                                                        int tid, bool isIssuer) {
#if HAVE_TDM
  if (isIssuer) {
    // K: 64 rows x 128 halves, row stride 128 elems; pad 64DW->+4DW => LDS stride 136 halves
    tdm_load_2d(lds_addr_of(sKbuf), (unsigned long long)(size_t)Kchunk,
                HEAD_D, K_TILE, HEAD_D, K_TILE, HEAD_D, /*64DW*/5u, /*4DW*/3u);
    // Vt: 128 rows x 64 halves, row stride SEQ_L elems; pad 32DW->+4DW => LDS stride 72 halves
    tdm_load_2d(lds_addr_of(sVbuf), (unsigned long long)(size_t)Vchunk,
                K_TILE, HEAD_D, K_TILE, HEAD_D, SEQ_L, /*32DW*/4u, /*4DW*/3u);
  }
#else
  for (int idx = tid; idx < K_TILE * HEAD_D / 8; idx += BLOCK_T) {
    int r = idx >> 4, c = idx & 15;
    ((uint4*)&sKbuf[r * QK_STRIDE])[c] = ((const uint4*)(Kchunk + (size_t)r * HEAD_D))[c];
  }
  for (int idx = tid; idx < HEAD_D * K_TILE / 8; idx += BLOCK_T) {
    int r = idx >> 3, c = idx & 7;
    ((uint4*)&sVbuf[r * V_STRIDE])[c] = ((const uint4*)(Vchunk + (size_t)r * SEQ_L))[c];
  }
#endif
}

// ================= pre-pass kernels =================
__global__ __launch_bounds__(256) void cvt_bf16_kernel(const float* __restrict__ in,
                                                       unsigned short* __restrict__ out, int n4) {
  int i = blockIdx.x * 256 + threadIdx.x;
  if (i < n4) pack4_store(&out[(size_t)i * 4], ((const float4*)in)[i]);
}

__global__ __launch_bounds__(256) void vtrans_bf16_kernel(const float* __restrict__ V,
                                                          unsigned short* __restrict__ Vt) {
  __shared__ unsigned short t[64][68];
  const int k0 = blockIdx.x * 64, f0 = blockIdx.y * 64, b = blockIdx.z;
  const float* src = V + ((size_t)b * SEQ_L + k0) * HEAD_D + f0;
  #pragma unroll
  for (int j = 0; j < 16; ++j) {
    int e = j * 256 + threadIdx.x;
    int r = e >> 6, c = e & 63;                 // key row r, feature col c (coalesced reads)
    t[r][c] = f2bf(src[(size_t)r * HEAD_D + c]);
  }
  __syncthreads();
  unsigned short* dst = Vt + ((size_t)b * HEAD_D + f0) * SEQ_L + k0;
  #pragma unroll
  for (int j = 0; j < 16; ++j) {
    int e = j * 256 + threadIdx.x;
    int f = e >> 6, k = e & 63;                 // coalesced writes along keys
    dst[(size_t)f * SEQ_L + k] = t[k][f];
  }
}

// ================= main flash-attention kernel =================
__global__ __launch_bounds__(BLOCK_T)
void fa_mulmask_kernel(const float* __restrict__ Q,
                       const unsigned short* __restrict__ Kbf,   // bf16 row-major [B][L][D]
                       const unsigned short* __restrict__ Vt,    // bf16 transposed [B][D][L]
                       const int* __restrict__ maskp,
                       float* __restrict__ Out) {
  __shared__ unsigned short sK [2][K_TILE * QK_STRIDE];
  __shared__ unsigned short sVt[2][HEAD_D * V_STRIDE];
  __shared__ unsigned short sP [4 * 16 * P_STRIDE];

  const int tid   = threadIdx.x;
  const int lane  = tid & 31;
  const int wid   = tid >> 5;
  const int col   = lane & 15;
  const int hi    = lane >> 4;
  const int b     = blockIdx.y;
  const int qBase = blockIdx.x * Q_TILE;

  // wave-uniform SGPR condition for TDM issue / tensorcnt wait (scalar branch, no EXEC mask)
  const bool isIssuer = (__builtin_amdgcn_readfirstlane(wid) == 0);

  const bool domask = (maskp[0] != 0);
  constexpr float NEG_BIG = 1.0f + 1000000000.0f;
  constexpr float MASKF   = 1.0f - NEG_BIG;          // ~ -1e9, reference fp32 math
  const float inv_den   = 1.0f / sqrtf((float)SEQ_L);
  const float fct_mask  = MASKF * inv_den;           // folded: one mul per score element
  const float fct_plain = inv_den;

  // ---- build Q A-fragments once, straight from global fp32 ----
  v16bf aQ[4];
  {
    const float* qrow = Q + ((size_t)b * SEQ_L + qBase + wid * 16 + col) * HEAD_D;
    #pragma unroll
    for (int ks = 0; ks < 4; ++ks) {
      union { v16bf v; unsigned short u[16]; } fa;
      const float* p0 = qrow + ks * 32 + hi * 8;
      const float* p1 = qrow + ks * 32 + 16 + hi * 8;
      float4 x0 = *(const float4*)(p0), x1 = *(const float4*)(p0 + 4);
      float4 y0 = *(const float4*)(p1), y1 = *(const float4*)(p1 + 4);
      fa.u[0]  = f2bf(x0.x); fa.u[1]  = f2bf(x0.y); fa.u[2]  = f2bf(x0.z); fa.u[3]  = f2bf(x0.w);
      fa.u[4]  = f2bf(x1.x); fa.u[5]  = f2bf(x1.y); fa.u[6]  = f2bf(x1.z); fa.u[7]  = f2bf(x1.w);
      fa.u[8]  = f2bf(y0.x); fa.u[9]  = f2bf(y0.y); fa.u[10] = f2bf(y0.z); fa.u[11] = f2bf(y0.w);
      fa.u[12] = f2bf(y1.x); fa.u[13] = f2bf(y1.y); fa.u[14] = f2bf(y1.z); fa.u[15] = f2bf(y1.w);
      aQ[ks] = fa.v;
    }
  }

  const v8f vzero = {0.f, 0.f, 0.f, 0.f, 0.f, 0.f, 0.f, 0.f};
  float m_i[8], l_i[8];
  v8f o[8];
  #pragma unroll
  for (int i = 0; i < 8; ++i) { m_i[i] = -3.0e38f; l_i[i] = 0.0f; }
  #pragma unroll
  for (int vt = 0; vt < 8; ++vt) o[vt] = vzero;

  unsigned short* pbase = &sP[wid * 16 * P_STRIDE];

  const unsigned short* Kb = Kbf + (size_t)b * SEQ_L * HEAD_D;
  const unsigned short* Vb = Vt  + (size_t)b * HEAD_D * SEQ_L;

  // prologue: start chunk 0 into buffer 0
  issue_chunk_load(Kb, Vb, sK[0], sVt[0], tid, isIssuer);

  // Multiplicative mask keeps masked entries in the softmax -> sweep ALL keys.
  for (int kc = 0; kc < N_KCHUNK; ++kc) {
    const int buf = kc & 1;
#if HAVE_TDM
    if (isIssuer) __builtin_amdgcn_s_wait_tensorcnt(0);   // chunk kc landed
#endif
    __syncthreads();   // chunk kc visible to all waves; buffer buf^1 free (iter kc-1 done)

    if (kc + 1 < N_KCHUNK)
      issue_chunk_load(Kb + (size_t)(kc + 1) * K_TILE * HEAD_D,
                       Vb + (size_t)(kc + 1) * K_TILE,
                       sK[buf ^ 1], sVt[buf ^ 1], tid, isIssuer);

    // ---- S = Q * K^T : 16 rows x 64 keys, f32 accum ----
    v8f s[4];
    #pragma unroll
    for (int nt = 0; nt < 4; ++nt) {
      v8f acc = vzero;
      const unsigned short* krow = &sK[buf][(nt * 16 + col) * QK_STRIDE];
      #pragma unroll
      for (int ks = 0; ks < 4; ++ks) {
        v16bf bK = load_frag_b(krow + ks * 32, hi);
        acc = __builtin_amdgcn_wmma_f32_16x16x32_bf16(false, aQ[ks], false, bK,
                                                      (short)0, acc, false, false);
      }
      s[nt] = acc;
    }

    // ---- multiplicative future mask (k >= q incl. diagonal) + 1/sqrt(Lk), folded ----
    #pragma unroll
    for (int nt = 0; nt < 4; ++nt) {
      int kk = kc * K_TILE + nt * 16 + col;
      #pragma unroll
      for (int i = 0; i < 8; ++i) {
        int qq = qBase + wid * 16 + i + 8 * hi;
        float fct = (domask && (kk >= qq)) ? fct_mask : fct_plain;
        s[nt][i] = s[nt][i] * fct;
      }
    }

    // ---- online softmax update ----
    #pragma unroll
    for (int i = 0; i < 8; ++i) {
      float mx = fmaxf(fmaxf(s[0][i], s[1][i]), fmaxf(s[2][i], s[3][i]));
      mx = fmaxf(mx, __shfl_xor(mx, 1));
      mx = fmaxf(mx, __shfl_xor(mx, 2));
      mx = fmaxf(mx, __shfl_xor(mx, 4));
      mx = fmaxf(mx, __shfl_xor(mx, 8));
      float mNew  = fmaxf(m_i[i], mx);
      float alpha = __expf(m_i[i] - mNew);
      m_i[i] = mNew;
      float ps = 0.0f;
      #pragma unroll
      for (int nt = 0; nt < 4; ++nt) {
        float p = __expf(s[nt][i] - mNew);
        s[nt][i] = p;
        ps += p;
      }
      ps += __shfl_xor(ps, 1);
      ps += __shfl_xor(ps, 2);
      ps += __shfl_xor(ps, 4);
      ps += __shfl_xor(ps, 8);
      l_i[i] = l_i[i] * alpha + ps;
      #pragma unroll
      for (int vt = 0; vt < 8; ++vt) o[vt][i] *= alpha;
    }

    // ---- stage P (C/D layout -> row-major bf16) in wave-private LDS ----
    #pragma unroll
    for (int nt = 0; nt < 4; ++nt) {
      #pragma unroll
      for (int i = 0; i < 8; ++i)
        pbase[(i + 8 * hi) * P_STRIDE + nt * 16 + col] = f2bf(s[nt][i]);
    }
    __builtin_amdgcn_wave_barrier();
    asm volatile("s_wait_dscnt 0x0" ::: "memory");
    __builtin_amdgcn_wave_barrier();

    // ---- O += P * V ----
    #pragma unroll
    for (int ks2 = 0; ks2 < 2; ++ks2) {
      v16bf aP = load_frag_a(&pbase[col * P_STRIDE + ks2 * 32], hi);
      #pragma unroll
      for (int vt = 0; vt < 8; ++vt) {
        v16bf bV = load_frag_b(&sVt[buf][(vt * 16 + col) * V_STRIDE + ks2 * 32], hi);
        o[vt] = __builtin_amdgcn_wmma_f32_16x16x32_bf16(false, aP, false, bV,
                                                        (short)0, o[vt], false, false);
      }
    }
  }

  // ---- normalize and store fp32 ----
  float* Ob = Out + ((size_t)b * SEQ_L + qBase + wid * 16) * HEAD_D;
  #pragma unroll
  for (int i = 0; i < 8; ++i) {
    float rinv = 1.0f / l_i[i];
    #pragma unroll
    for (int vt = 0; vt < 8; ++vt)
      Ob[(size_t)(i + 8 * hi) * HEAD_D + vt * 16 + col] = o[vt][i] * rinv;
  }
}

extern "C" void kernel_launch(void* const* d_in, const int* in_sizes, int n_in,
                              void* d_out, int out_size, void* d_ws, size_t ws_size,
                              hipStream_t stream) {
  const float* Q = (const float*)d_in[0];
  const float* K = (const float*)d_in[1];
  const float* V = (const float*)d_in[2];
  const int*   m = (const int*)d_in[3];
  const int B = in_sizes[0] / (SEQ_L * HEAD_D);     // 16
  const size_t elems = (size_t)B * SEQ_L * HEAD_D;  // 4.19M; ws need = 2*elems*2B ~ 16.8MB

  unsigned short* Kbf = (unsigned short*)d_ws;      // bf16 row-major K
  unsigned short* Vtr = Kbf + elems;                // bf16 transposed V [B][D][L]

  cvt_bf16_kernel<<<(int)(elems / 4 + 255) / 256, 256, 0, stream>>>(K, Kbf, (int)(elems / 4));
  vtrans_bf16_kernel<<<dim3(SEQ_L / 64, HEAD_D / 64, B), 256, 0, stream>>>(V, Vtr);

  dim3 grid(SEQ_L / Q_TILE, B);                     // 32 x 16 = 512 workgroups
  fa_mulmask_kernel<<<grid, BLOCK_T, 0, stream>>>(Q, Kbf, Vtr, m, (float*)d_out);
}
